// MixtureOfExperts_50105088475463
// MI455X (gfx1250) — matrix operand order repaired and checked by reading, not compiled, not run on hardware
//
#include <hip/hip_runtime.h>
#include <hip/hip_bf16.h>

// Problem constants (from reference): H=2048, E=8, K=2, N=4096
#define HDIM 2048
#define NEXP 8
#define NTOK 4096
#define TOPK 2

// GEMM tiling: block = 256 threads (8 wave32), tile M=128, N=64, Kstep=32
#define BM 128
#define BN 64
#define BK 32
#define ASTRIDE (BK + 16)   // 48 u16 = 96B row stride (16B aligned)

typedef __attribute__((ext_vector_type(16))) __bf16 v16bf;
typedef __attribute__((ext_vector_type(8)))  float  v8f;

// CDNA5 async global->LDS path (ASYNCcnt), with sync fallback
#if defined(__has_builtin)
#if __has_builtin(__builtin_amdgcn_global_load_async_to_lds_b128) && \
    __has_builtin(__builtin_amdgcn_s_wait_asynccnt)
#define USE_ASYNC_LDS 1
#endif
#endif
#ifndef USE_ASYNC_LDS
#define USE_ASYNC_LDS 0
#endif

__device__ __forceinline__ unsigned short f2bf(float f) {
    unsigned int u = __float_as_uint(f);
    // round-to-nearest-even bf16
    return (unsigned short)((u + 0x7FFFu + ((u >> 16) & 1u)) >> 16);
}

#if USE_ASYNC_LDS
// builtin signature (from clang diagnostic): (v4i as1*, v4i as3*, imm, imm)
typedef int v4i_vs __attribute__((vector_size(16)));
typedef __attribute__((address_space(1))) v4i_vs* gv4i_p;
typedef __attribute__((address_space(3))) v4i_vs* lv4i_p;

__device__ __forceinline__ void async_copy16(void* lds, const void* gsrc) {
    __builtin_amdgcn_global_load_async_to_lds_b128((gv4i_p)(gsrc), (lv4i_p)(lds), 0, 0);
}
#endif

// ---------------------------------------------------------------------------
// small helpers
// ---------------------------------------------------------------------------
__global__ void zero_counts_kernel(unsigned int* counts) {
    if (threadIdx.x < NEXP) counts[threadIdx.x] = 0u;
}

__global__ void zero_f32_kernel(float4* __restrict__ p, int n4) {
    int i = blockIdx.x * blockDim.x + threadIdx.x;
    int s = gridDim.x * blockDim.x;
    float4 z = make_float4(0.f, 0.f, 0.f, 0.f);
    for (; i < n4; i += s) p[i] = z;
}

// f32 -> bf16 row-major pack (activations)
__global__ void cvt_bf16_kernel(const float4* __restrict__ src, uint2* __restrict__ dst, int n4) {
    int i = blockIdx.x * blockDim.x + threadIdx.x;
    int s = gridDim.x * blockDim.x;
    for (; i < n4; i += s) {
        float4 v = src[i];
        uint2 o;
        o.x = (unsigned)f2bf(v.x) | ((unsigned)f2bf(v.y) << 16);
        o.y = (unsigned)f2bf(v.z) | ((unsigned)f2bf(v.w) << 16);
        dst[i] = o;
    }
}

// f32 weights [E][H][H] (k-major rows) -> bf16 pre-swizzled into the WMMA B
// layout: wpk[e][p][n] : u32 = ( bf16(W[e][2p][n]) , bf16(W[e][2p+1][n]) )
// so the GEMM B-stage is a pure 16-byte copy with zero in-loop VALU.
__global__ void cvt_w_pack_kernel(const float* __restrict__ W, unsigned int* __restrict__ wpk) {
    int idx = blockIdx.x * blockDim.x + threadIdx.x;   // one thread per 4 n
    int n4  = idx & (HDIM / 4 - 1);
    int rst = idx >> 9;                                // / (H/4)
    int p   = rst & (HDIM / 2 - 1);
    int e   = rst >> 10;                               // / (H/2)
    if (e >= NEXP) return;
    const float* r0 = W + ((size_t)e * HDIM + 2 * p) * HDIM + n4 * 4;
    float4 a = *(const float4*)r0;
    float4 b = *(const float4*)(r0 + HDIM);
    uint4 o;
    o.x = (unsigned)f2bf(a.x) | ((unsigned)f2bf(b.x) << 16);
    o.y = (unsigned)f2bf(a.y) | ((unsigned)f2bf(b.y) << 16);
    o.z = (unsigned)f2bf(a.z) | ((unsigned)f2bf(b.z) << 16);
    o.w = (unsigned)f2bf(a.w) | ((unsigned)f2bf(b.w) << 16);
    *(uint4*)(wpk + ((size_t)e * (HDIM / 2) + p) * HDIM + n4 * 4) = o;
}

// ---------------------------------------------------------------------------
// gate: logits -> softmax -> top-2, one wave32 per token
// ---------------------------------------------------------------------------
__global__ __launch_bounds__(256) void gate_kernel(
    const float* __restrict__ x, const float* __restrict__ gW, const float* __restrict__ gb,
    float* __restrict__ gate_out,                      // [N, E]
    int* __restrict__ eidx, float* __restrict__ ewgt,  // [N*2]
    unsigned int* __restrict__ counts)
{
    const int wave = threadIdx.x >> 5;
    const int lane = threadIdx.x & 31;
    const int n = blockIdx.x * 8 + wave;
    if (n >= NTOK) return;

    float acc[NEXP];
#pragma unroll
    for (int e = 0; e < NEXP; ++e) acc[e] = 0.f;

    const float* xr = x + (size_t)n * HDIM;
    for (int h = lane; h < HDIM; h += 32) {
        float xv = xr[h];
#pragma unroll
        for (int e = 0; e < NEXP; ++e) acc[e] += xv * gW[e * HDIM + h];
    }
#pragma unroll
    for (int e = 0; e < NEXP; ++e) {
        float v = acc[e];
#pragma unroll
        for (int off = 16; off > 0; off >>= 1) v += __shfl_xor(v, off, 32);
        acc[e] = v + gb[e];
    }
    if (lane == 0) {
        float mx = acc[0];
#pragma unroll
        for (int e = 1; e < NEXP; ++e) mx = fmaxf(mx, acc[e]);
        float s[NEXP], sum = 0.f;
#pragma unroll
        for (int e = 0; e < NEXP; ++e) { s[e] = __expf(acc[e] - mx); sum += s[e]; }
        float inv = 1.0f / sum;
        int i0 = 0;
#pragma unroll
        for (int e = 0; e < NEXP; ++e) {
            s[e] *= inv;
            gate_out[n * NEXP + e] = s[e];
            if (s[e] > s[i0]) i0 = e;          // lowest index wins ties (matches top_k)
        }
        int i1 = (i0 == 0) ? 1 : 0;
#pragma unroll
        for (int e = 0; e < NEXP; ++e)
            if (e != i0 && s[e] > s[i1]) i1 = e;

        eidx[2 * n + 0] = i0;  ewgt[2 * n + 0] = s[i0];
        eidx[2 * n + 1] = i1;  ewgt[2 * n + 1] = s[i1];
        atomicAdd(&counts[i0], 1u);
        atomicAdd(&counts[i1], 1u);
    }
}

__global__ void scan_kernel(const unsigned int* __restrict__ counts,
                            unsigned int* __restrict__ offsets,
                            unsigned int* __restrict__ cursor) {
    if (threadIdx.x == 0 && blockIdx.x == 0) {
        unsigned int run = 0;
        for (int e = 0; e < NEXP; ++e) { offsets[e] = run; cursor[e] = run; run += counts[e]; }
    }
}

__global__ void scatter_kernel(const int* __restrict__ eidx, const float* __restrict__ ewgt,
                               unsigned int* __restrict__ cursor,
                               int* __restrict__ row_token, float* __restrict__ row_wgt) {
    int i = blockIdx.x * blockDim.x + threadIdx.x;
    if (i >= NTOK * TOPK) return;
    int e = eidx[i];
    unsigned int pos = atomicAdd(&cursor[e], 1u);
    row_token[pos] = i >> 1;
    row_wgt[pos]   = ewgt[i];
}

// ---------------------------------------------------------------------------
// grouped GEMM over selected (token, expert) rows, bf16 WMMA, f32 accumulate
// LAYER==1: H1 = relu(Xe @ W1[e] + b1[e])  -> h1out (bf16)
// LAYER==2: out[token] += w * (H1e @ W2[e] + b2[e])  (atomic f32)
// B side is pre-swizzled (wpk) so staging is a pure 16B copy per thread.
// ---------------------------------------------------------------------------
template <int LAYER>
__global__ __launch_bounds__(256) void moe_gemm_kernel(
    const unsigned short* __restrict__ Abase,   // L1: xbf [NTOK,H]; L2: h1bf [N*K, H]
    const unsigned int* __restrict__ Wpk,       // [E][H/2][H] u32 K-pair packed
    const float* __restrict__ bias,             // [E, H]
    const unsigned int* __restrict__ counts,
    const unsigned int* __restrict__ offsets,
    const int* __restrict__ row_token,
    const float* __restrict__ row_wgt,
    unsigned short* __restrict__ h1out,
    float* __restrict__ aggout)
{
    const int e = blockIdx.z;
    const unsigned int Me = counts[e];
    const int m0 = blockIdx.y * BM;
    if ((unsigned)m0 >= Me) return;                      // uniform per block
    const int n0 = blockIdx.x * BN;
    const unsigned int off = offsets[e];

#if USE_ASYNC_LDS
    __shared__ __align__(16) unsigned short As[2][BM][ASTRIDE];
    __shared__ __align__(16) unsigned int   Bs[2][BK / 2][BN];
#else
    __shared__ __align__(16) unsigned short As[1][BM][ASTRIDE];
    __shared__ __align__(16) unsigned int   Bs[1][BK / 2][BN];
#endif

    const int tid = threadIdx.x;

    // A loader: thread stages rows (tid>>2) and (tid>>2)+64, cols (tid&3)*8..+7.
    // Invalid rows are clamped to row 0 (always-valid global load); the garbage
    // only lands in C rows >= Me which the epilogue masks out.
    const int arow0 = tid >> 2;
    const int acol  = (tid & 3) * 8;
    size_t rowA0 = 0, rowA1 = 0;
    {
        int m = m0 + arow0;
        if ((unsigned)m < Me)
            rowA0 = (LAYER == 1) ? (size_t)row_token[off + m] : (size_t)(off + m);
        m = m0 + arow0 + 64;
        if ((unsigned)m < Me)
            rowA1 = (LAYER == 1) ? (size_t)row_token[off + m] : (size_t)(off + m);
    }
    const unsigned short* aptr0 = Abase + rowA0 * HDIM + acol;
    const unsigned short* aptr1 = Abase + rowA1 * HDIM + acol;

    // B loader: thread stages pair-row (tid>>4), u32 cols (tid&15)*4..+3
    const int bp  = tid >> 4;                  // 0..15
    const int bnc = (tid & 15) * 4;            // u32 column
    const unsigned int* bptr =
        Wpk + (size_t)e * (HDIM / 2) * HDIM + (size_t)bp * HDIM + n0 + bnc;

    const int wv    = tid >> 5;
    const int lane  = tid & 31;
    const int wm    = wv & 3;        // 4 m-waves -> rows wm*32 .. +31
    const int wn    = wv >> 2;       // 2 n-waves -> cols wn*32 .. +31
    const int lhalf = lane >> 4;     // lane half per ISA VGPR layout
    const int l16   = lane & 15;

    v8f acc[2][2];
#pragma unroll
    for (int i = 0; i < 2; ++i)
#pragma unroll
        for (int j = 0; j < 2; ++j)
#pragma unroll
            for (int r = 0; r < 8; ++r) acc[i][j][r] = 0.0f;

    const int NK = HDIM / BK;        // 64 K-steps

    auto stage = [&](int ks, int b) {
        const unsigned short* a0 = aptr0 + ks * BK;
        const unsigned short* a1 = aptr1 + ks * BK;
        const unsigned int*   bw = bptr + (size_t)ks * (BK / 2) * HDIM;
#if USE_ASYNC_LDS
        async_copy16(&As[b][arow0][acol],      a0);
        async_copy16(&As[b][arow0 + 64][acol], a1);
        async_copy16(&Bs[b][bp][bnc],          bw);
#else
        *(uint4*)&As[b][arow0][acol]      = *(const uint4*)a0;
        *(uint4*)&As[b][arow0 + 64][acol] = *(const uint4*)a1;
        *(uint4*)&Bs[b][bp][bnc]          = *(const uint4*)bw;
#endif
    };

    auto compute = [&](int b) {
        union Frag { unsigned int u[8]; v16bf v; } afrag[2], bfrag[2];
#pragma unroll
        for (int mi = 0; mi < 2; ++mi) {
            int mrow = wm * 32 + mi * 16 + l16;
#pragma unroll
            for (int v = 0; v < 8; ++v) {
                // A 16x32 bf16: dword v holds K = (v/4)*16 + half*8 + (v%4)*2 (+1)
                int kk = (v >> 2) * 16 + lhalf * 8 + (v & 3) * 2;
                afrag[mi].u[v] = *(const unsigned int*)&As[b][mrow][kk];
            }
        }
#pragma unroll
        for (int ni = 0; ni < 2; ++ni) {
            int ncol = wn * 32 + ni * 16 + l16;
#pragma unroll
            for (int v = 0; v < 8; ++v) {
                // B 32x16 bf16: dword v holds K pair ((v/4)*8 + half*4 + v%4) for column l16
                bfrag[ni].u[v] = Bs[b][(v >> 2) * 8 + lhalf * 4 + (v & 3)][ncol];
            }
        }
#pragma unroll
        for (int mi = 0; mi < 2; ++mi)
#pragma unroll
            for (int ni = 0; ni < 2; ++ni)
                acc[mi][ni] = __builtin_amdgcn_wmma_f32_16x16x32_bf16(
                    false, afrag[mi].v, false, bfrag[ni].v,
                    (short)0, acc[mi][ni], false, false);
    };

#if USE_ASYNC_LDS
    // double-buffered async pipeline: one barrier + s_wait_asynccnt per K-step
    stage(0, 0);
    int buf = 0;
    for (int ks = 0; ks < NK; ++ks) {
        __builtin_amdgcn_s_wait_asynccnt(0);
        __syncthreads();
        if (ks + 1 < NK) stage(ks + 1, buf ^ 1);
        compute(buf);
        buf ^= 1;
    }
#else
    for (int ks = 0; ks < NK; ++ks) {
        __syncthreads();
        stage(ks, 0);
        __syncthreads();
        compute(0);
    }
#endif

    // ---- epilogue ----
    const float* brow = bias + (size_t)e * HDIM;
#pragma unroll
    for (int mi = 0; mi < 2; ++mi) {
#pragma unroll
        for (int ni = 0; ni < 2; ++ni) {
            int ng = n0 + wn * 32 + ni * 16 + l16;
            float bv = brow[ng];
#pragma unroll
            for (int r = 0; r < 8; ++r) {
                // C 16x16 f32: VGPR r holds M = r + half*8
                int m = m0 + wm * 32 + mi * 16 + lhalf * 8 + r;
                if ((unsigned)m < Me) {
                    float val = acc[mi][ni][r] + bv;
                    if (LAYER == 1) {
                        val = fmaxf(val, 0.0f);
                        h1out[(size_t)(off + m) * HDIM + ng] = f2bf(val);
                    } else {
                        int   tok = row_token[off + m];
                        float w   = row_wgt[off + m];
                        atomicAdd(&aggout[(size_t)tok * HDIM + ng], w * val);
                    }
                }
            }
        }
    }
}

// ---------------------------------------------------------------------------
// launch
// ---------------------------------------------------------------------------
extern "C" void kernel_launch(void* const* d_in, const int* in_sizes, int n_in,
                              void* d_out, int out_size, void* d_ws, size_t ws_size,
                              hipStream_t stream) {
    (void)in_sizes; (void)n_in; (void)out_size; (void)ws_size;

    const float* x   = (const float*)d_in[0];
    const float* gW  = (const float*)d_in[1];
    const float* gb  = (const float*)d_in[2];
    const float* W1  = (const float*)d_in[3];
    const float* b1  = (const float*)d_in[4];
    const float* W2  = (const float*)d_in[5];
    const float* b2  = (const float*)d_in[6];

    float* out_agg  = (float*)d_out;                   // [N, H]
    float* out_gate = out_agg + (size_t)NTOK * HDIM;   // [N, E]

    // ---- workspace layout ----
    char* ws = (char*)d_ws;
    size_t o = 0;
    auto take = [&](size_t bytes) { char* p = ws + o; o = (o + bytes + 255) & ~(size_t)255; return p; };
    unsigned int* counts  = (unsigned int*)take(NEXP * sizeof(unsigned int));
    unsigned int* offsets = (unsigned int*)take(NEXP * sizeof(unsigned int));
    unsigned int* cursor  = (unsigned int*)take(NEXP * sizeof(unsigned int));
    int*    eidx      = (int*)  take((size_t)NTOK * TOPK * sizeof(int));
    float*  ewgt      = (float*)take((size_t)NTOK * TOPK * sizeof(float));
    int*    row_token = (int*)  take((size_t)NTOK * TOPK * sizeof(int));
    float*  row_wgt   = (float*)take((size_t)NTOK * TOPK * sizeof(float));
    unsigned short* xbf  = (unsigned short*)take((size_t)NTOK * HDIM * 2);
    unsigned short* h1bf = (unsigned short*)take((size_t)NTOK * TOPK * HDIM * 2);
    unsigned int*   wpk1 = (unsigned int*)take((size_t)NEXP * HDIM * HDIM * 2);
    unsigned int*   wpk2 = (unsigned int*)take((size_t)NEXP * HDIM * HDIM * 2);

    // 1) init
    zero_counts_kernel<<<1, 32, 0, stream>>>(counts);
    zero_f32_kernel<<<2048, 256, 0, stream>>>((float4*)out_agg, NTOK * HDIM / 4);

    // 2) f32 -> bf16 conversions; weights go out pre-swizzled for the WMMA B
    //    layout (134 MB bf16 total: L2-resident for the GEMMs)
    cvt_bf16_kernel<<<2048, 256, 0, stream>>>((const float4*)x, (uint2*)xbf, NTOK * HDIM / 4);
    {
        int witems = NEXP * (HDIM / 2) * (HDIM / 4);   // one thread per 4 n
        cvt_w_pack_kernel<<<witems / 256, 256, 0, stream>>>(W1, wpk1);
        cvt_w_pack_kernel<<<witems / 256, 256, 0, stream>>>(W2, wpk2);
    }

    // 3) gate + routing
    gate_kernel<<<NTOK / 8, 256, 0, stream>>>(x, gW, gb, out_gate, eidx, ewgt, counts);
    scan_kernel<<<1, 32, 0, stream>>>(counts, offsets, cursor);
    scatter_kernel<<<(NTOK * TOPK + 255) / 256, 256, 0, stream>>>(eidx, ewgt, cursor, row_token, row_wgt);

    // 4) grouped expert GEMMs (bf16 WMMA), tiles beyond each expert's count exit
    dim3 ggrid(HDIM / BN, NTOK / BM, NEXP);   // (32, 32, 8)
    moe_gemm_kernel<1><<<ggrid, 256, 0, stream>>>(xbf,  wpk1, b1, counts, offsets,
                                                  row_token, row_wgt, h1bf, nullptr);
    moe_gemm_kernel<2><<<ggrid, 256, 0, stream>>>(h1bf, wpk2, b2, counts, offsets,
                                                  row_token, row_wgt, nullptr, out_agg);
}
